// UResNetDecoder_77257871720758
// MI455X (gfx1250) — compile-verified, matrix operand order checked by reading, not run on hardware
//
#include <hip/hip_runtime.h>

// ---------------------------------------------------------------------------
// CDNA5 (gfx1250) UResNet decoder, v3.
// Implicit-GEMM 3x3x3 convs via V_WMMA_F32_16X16X32_BF16:
//  - activations bf16 channels-last [B, voxel, Cpad] -> B operand is two
//    contiguous 16B global loads per lane
//  - weight tile (16 Cout rows x Kpad) staged in LDS per 4-wave workgroup,
//    A operand fed by ds_load_b128 (global weight traffic = once per block)
//  - all index math via shifts (S is a power of two), incremental tap decode,
//    3-D grids: no per-thread integer divisions in hot paths
//  - fused BN / residual / ReLU epilogue, dual f32 + bf16 outputs
// ---------------------------------------------------------------------------

typedef __attribute__((ext_vector_type(16))) __bf16 v16bf;
typedef __attribute__((ext_vector_type(8)))  __bf16 v8bf;
typedef __attribute__((ext_vector_type(8)))  float  v8f;

#define BN_EPS 1e-5f

// ---------------- elementwise helpers -------------------------------------

// grid: (cdiv(DHW,256), B*C)
__global__ __launch_bounds__(256)
void urd_bn_relu_k(const float* __restrict__ x,
                   const float* __restrict__ g, const float* __restrict__ b,
                   const float* __restrict__ m, const float* __restrict__ v,
                   float* __restrict__ y, int C, int DHW) {
    int vox = blockIdx.x * blockDim.x + threadIdx.x;
    if (vox >= DHW) return;
    int bc = blockIdx.y;
    int bb = bc / C;                 // uniform scalar div, once per block
    int c  = bc - bb * C;
    float inv = g[c] * rsqrtf(v[c] + BN_EPS);
    size_t i = (size_t)bc * DHW + vox;
    float val = x[i] * inv + (b[c] - m[c] * inv);
    y[i] = fmaxf(val, 0.0f);
}

// encoder skip (f32 channel-major) -> bf16 channels-last concat, ch [0,Cenc)
// grid: (cdiv(DHW,256), B*Cenc)
__global__ __launch_bounds__(256)
void urd_copy_enc_k(const float* __restrict__ enc, __bf16* __restrict__ xcat,
                    int Cenc, int Ctot, int DHW) {
    int vox = blockIdx.x * blockDim.x + threadIdx.x;
    if (vox >= DHW) return;
    int bc = blockIdx.y;
    int bb = bc / Cenc;
    int c  = bc - bb * Cenc;
    xcat[((size_t)bb * DHW + vox) * Ctot + c] = (__bf16)enc[(size_t)bc * DHW + vox];
}

// transposed conv k=2 s=2 -> bf16 channels-last concat, ch [coff, coff+Cout)
// grid: (cdiv(DHWo,256), B*Cout)
__global__ __launch_bounds__(256)
void urd_convt2_k(const float* __restrict__ in, const float* __restrict__ wt,
                  __bf16* __restrict__ xcat, int Cin, int Cout, int Ctot,
                  int coff, int logSin) {
    int logSo = logSin + 1;
    int DHWo  = 1 << (3 * logSo);
    int vox = blockIdx.x * blockDim.x + threadIdx.x;
    if (vox >= DHWo) return;
    int bc = blockIdx.y;
    int bb = bc / Cout;
    int co = bc - bb * Cout;
    int Smo = (1 << logSo) - 1;
    int z = vox >> (2 * logSo);
    int y = (vox >> logSo) & Smo;
    int x = vox & Smo;
    int zi = z >> 1, kd = z & 1;
    int yi = y >> 1, kh = y & 1;
    int xi = x >> 1, kw = x & 1;
    int DHWi = 1 << (3 * logSin);
    const float* ip = in + (size_t)bb * Cin * DHWi +
                      (((size_t)zi << (2 * logSin)) + ((size_t)yi << logSin) + xi);
    const float* wp = wt + (size_t)((co * 2 + kd) * 2 + kh) * 2 + kw;
    float s = 0.0f;
    for (int ci = 0; ci < Cin; ++ci)
        s += ip[(size_t)ci * DHWi] * wp[(size_t)ci * Cout * 8];
    xcat[((size_t)bb * DHWo + vox) * Ctot + coff + co] = (__bf16)s;
}

// zero the 16 padding channels [C0, C0+16) of a channels-last bf16 buffer
// (only launched when CP - C0 == 16)
__global__ __launch_bounds__(256)
void urd_padzero_k(__bf16* __restrict__ act, int C0, int CP, long total) {
    long i = (long)blockIdx.x * blockDim.x + threadIdx.x;
    if (i >= total) return;
    long vox = i >> 4;
    int  c   = C0 + (int)(i & 15);
    act[vox * CP + c] = (__bf16)0.0f;
}

// f32 weights [Cout, Cin, ktaps] -> bf16 [Cout, ktaps, Cpad] (tap-major, 0-pad)
// grid: (1, Cout*ktaps), block: Cpad threads
__global__ __launch_bounds__(128)
void urd_wprep_k(const float* __restrict__ w, __bf16* __restrict__ wbf,
                 int Cin, int Cpad, int ktaps) {
    int yb  = blockIdx.y;
    int co  = yb / ktaps;            // uniform scalar div, once per block
    int tap = yb - co * ktaps;
    int ci  = threadIdx.x;
    float val = (ci < Cin) ? w[((size_t)co * Cin + ci) * ktaps + tap] : 0.0f;
    wbf[((size_t)co * ktaps + tap) * Cpad + ci] = (__bf16)val;
}

// ---------------- implicit-GEMM conv (WMMA) -------------------------------
// act: bf16 channels-last [B, DHW, CPAD] (zero-padded channels)
// wA : bf16 [Cout, KTAPS, CPAD]
// Block = 4 waves sharing one 16-row Cout tile over 256 voxels; the 16 x Kpad
// weight rows are staged to LDS once per block when they fit in 60 KB.
// grid: (DHW/256, Cout/16, batch)   (DHW is always a multiple of 256)
template <int CPAD, int KTAPS>
__global__ __launch_bounds__(128)
void urd_conv_wmma_k(const __bf16* __restrict__ act, const __bf16* __restrict__ wA,
                     const float* __restrict__ bng, const float* __restrict__ bnb,
                     const float* __restrict__ bnm, const float* __restrict__ bnv,
                     const float* __restrict__ resid,
                     float* __restrict__ outF, __bf16* __restrict__ outH,
                     int Cout, int CoutPad, int logS, int do_relu) {
    constexpr int  KPAD  = KTAPS * CPAD;
    constexpr bool STAGE = (16 * KPAD * 2) <= (60 * 1024);
    constexpr int  SMEMN = STAGE ? 16 * KPAD : 8;
    __shared__ __bf16 asmem[SMEMN];

    const int S    = 1 << logS;
    const int Smk  = S - 1;
    const int DHW  = 1 << (3 * logS);
    const int bb   = blockIdx.z;
    const int mt   = blockIdx.y;
    const int wid  = threadIdx.x >> 5;
    const int lane = threadIdx.x & 31;
    const int lo16 = lane & 15;
    const int hi   = lane >> 4;
    const int co0  = mt << 4;

    if constexpr (STAGE) {
        // 16 weight rows are contiguous in wA: a straight vector memcpy
        const v8bf* src = (const v8bf*)(wA + (size_t)co0 * KPAD);
        v8bf* dst = (v8bf*)asmem;
        constexpr int NV = 16 * KPAD / 8;
#pragma unroll 1
        for (int u = threadIdx.x; u < NV; u += 128) dst[u] = src[u];
        __syncthreads();
    }

    int vox[4], z[4], y[4], x[4];
#pragma unroll
    for (int s = 0; s < 4; ++s) {
        vox[s] = (blockIdx.x << 8) + (wid << 6) + (s << 4) + lo16;
        z[s] = vox[s] >> (2 * logS);
        y[s] = (vox[s] >> logS) & Smk;
        x[s] = vox[s] & Smk;
    }

    const __bf16* wrow = wA + (size_t)(co0 + lo16) * KPAD + (hi << 3);
    const __bf16* lrow = asmem + lo16 * KPAD + (hi << 3);
    const size_t actB = (size_t)bb * DHW;
    const v8bf zv = {};

    v8f acc[4] = {};
    int dz = (KTAPS == 27) ? -1 : 0;
    int dy = dz, dx = dz;
#pragma unroll 1
    for (int tap = 0; tap < KTAPS; ++tap) {
        bool ok[4];
        const __bf16* brow[4];
#pragma unroll
        for (int s = 0; s < 4; ++s) {
            int zz = z[s] + dz, yy = y[s] + dy, xx = x[s] + dx;
            ok[s] = ((unsigned)zz < (unsigned)S) & ((unsigned)yy < (unsigned)S) &
                    ((unsigned)xx < (unsigned)S);
            long sp = ((long)((zz << logS) + yy) << logS) + xx;
            brow[s] = act + (actB + sp) * CPAD + (hi << 3);
        }
#pragma unroll
        for (int cib = 0; cib < CPAD; cib += 32) {
            v16bf a;
            if constexpr (STAGE) {
                v8bf alo = *(const v8bf*)(lrow + tap * CPAD + cib);
                v8bf ahi = *(const v8bf*)(lrow + tap * CPAD + cib + 16);
                a = __builtin_shufflevector(alo, ahi,
                    0, 1, 2, 3, 4, 5, 6, 7, 8, 9, 10, 11, 12, 13, 14, 15);
            } else {
                v8bf alo = *(const v8bf*)(wrow + tap * CPAD + cib);
                v8bf ahi = *(const v8bf*)(wrow + tap * CPAD + cib + 16);
                a = __builtin_shufflevector(alo, ahi,
                    0, 1, 2, 3, 4, 5, 6, 7, 8, 9, 10, 11, 12, 13, 14, 15);
            }
#pragma unroll
            for (int s = 0; s < 4; ++s) {
                v8bf blo = ok[s] ? *(const v8bf*)(brow[s] + cib)      : zv;
                v8bf bhi = ok[s] ? *(const v8bf*)(brow[s] + cib + 16) : zv;
                v16bf bm = __builtin_shufflevector(blo, bhi,
                    0, 1, 2, 3, 4, 5, 6, 7, 8, 9, 10, 11, 12, 13, 14, 15);
                acc[s] = __builtin_amdgcn_wmma_f32_16x16x32_bf16(
                    false, a, false, bm, (short)0, acc[s], false, false);
            }
        }
        if (KTAPS == 27) {           // incremental 3x3x3 tap decode, no division
            if (++dx == 2) { dx = -1; if (++dy == 2) { dy = -1; ++dz; } }
        }
    }

#pragma unroll
    for (int r = 0; r < 8; ++r) {
        int co = co0 + r + (hi << 3);          // C/D layout: M = r + 8*(lane/16)
        float sc = 1.0f, sh = 0.0f;
        if (bng) {
            sc = bng[co] * rsqrtf(bnv[co] + BN_EPS);
            sh = bnb[co] - bnm[co] * sc;
        }
        size_t chbase = ((size_t)(bb * Cout + co)) << (3 * logS);
#pragma unroll
        for (int s = 0; s < 4; ++s) {
            float val = acc[s][r] * sc + sh;
            if (resid) val += resid[chbase + vox[s]];
            if (do_relu) val = fmaxf(val, 0.0f);
            if (outF) outF[chbase + vox[s]] = val;
            if (outH) outH[(actB + vox[s]) * CoutPad + co] = (__bf16)val;
        }
    }
}

// ---------------- host orchestration --------------------------------------

struct UrdDec { const float *g, *b, *m, *v, *wt; };
struct UrdBlk { const float *w1, *g1, *b1, *m1, *v1, *w2, *g2, *b2, *m2, *v2, *res_w; };

static inline int urd_cdiv(long a, long b) { return (int)((a + b - 1) / b); }

extern "C" void kernel_launch(void* const* d_in, const int* in_sizes, int n_in,
                              void* d_out, int out_size, void* d_ws, size_t ws_size,
                              hipStream_t stream) {
    const int PL[5] = {16, 32, 48, 64, 80};
    const long out_off[4] = {0, 65536, 458752, 2555904};

    auto F = [&](int i) { return (const float*)d_in[i]; };

    const float* enc[5];
    const float* xfinal;
    UrdDec dec[4];
    UrdBlk blk[4][2];

    for (int i = 0; i < 5; ++i) enc[i] = F(i);
    xfinal = F(5);

    // jax pytree flatten sorts dict keys ('blocks' < 'dec', sorted inside dicts);
    // fall back to insertion order if leaf sizes say otherwise.
    bool sorted_order = (n_in > 6 && in_sizes[6] == 64);
    if (sorted_order) {
        for (int L = 0; L < 4; ++L) {
            int b0 = 6 + 21 * L;   // b1,b2,g1,g2,m1,m2,res_w,v1,v2,w1,w2
            blk[L][0] = { F(b0+9), F(b0+2), F(b0+0), F(b0+4), F(b0+7),
                          F(b0+10), F(b0+3), F(b0+1), F(b0+5), F(b0+8), F(b0+6) };
            int b1 = b0 + 11;      // b1,b2,g1,g2,m1,m2,v1,v2,w1,w2
            blk[L][1] = { F(b1+8), F(b1+2), F(b1+0), F(b1+4), F(b1+6),
                          F(b1+9), F(b1+3), F(b1+1), F(b1+5), F(b1+7), nullptr };
        }
        for (int L = 0; L < 4; ++L) {
            int d0 = 90 + 5 * L;   // b,g,m,v,wt
            dec[L] = { F(d0+1), F(d0+0), F(d0+2), F(d0+3), F(d0+4) };
        }
    } else {
        for (int L = 0; L < 4; ++L) {
            int d0 = 6 + 5 * L;    // g,b,m,v,wt
            dec[L] = { F(d0+0), F(d0+1), F(d0+2), F(d0+3), F(d0+4) };
        }
        for (int L = 0; L < 4; ++L) {
            int b0 = 26 + 21 * L;  // w1,g1,b1,m1,v1,w2,g2,b2,m2,v2,res_w
            blk[L][0] = { F(b0+0), F(b0+1), F(b0+2), F(b0+3), F(b0+4),
                          F(b0+5), F(b0+6), F(b0+7), F(b0+8), F(b0+9), F(b0+10) };
            int b1 = b0 + 11;
            blk[L][1] = { F(b1+0), F(b1+1), F(b1+2), F(b1+3), F(b1+4),
                          F(b1+5), F(b1+6), F(b1+7), F(b1+8), F(b1+9), nullptr };
        }
    }

    // workspace carve
    char* wsb = (char*)d_ws;
    size_t off = 0;
    auto carve = [&](size_t bytes) -> void* {
        void* p = wsb + off;
        off += (bytes + 255) & ~(size_t)255;
        return p;
    };
    float*  bnx   = (float*)carve((size_t)2097152 * 4);    // pre-decoder bn/relu
    __bf16* xcatH = (__bf16*)carve((size_t)16777216 * 2);  // concat, ch-last bf16
    __bf16* t1h   = (__bf16*)carve((size_t)16777216 * 2);  // inter-conv, ch-last bf16
    __bf16* xbH   = (__bf16*)carve((size_t)16777216 * 2);  // block0 out, ch-last bf16
    float*  xbF   = (float*)carve((size_t)8388608 * 4);    // block0 out, f32 (residual)
    float*  resb  = (float*)carve((size_t)8388608 * 4);    // 1x1 residual, f32
    __bf16* wbf   = (__bf16*)carve((size_t)262144 * 2);    // bf16 weight staging
    (void)ws_size; (void)out_size;

    float* outf = (float*)d_out;

    auto launch_conv = [&](int CP, int KT, const __bf16* act, const float* w,
                           int Cin, int Cout, int CoutPad, int logS,
                           const float* g, const float* b, const float* m,
                           const float* v, const float* resid,
                           float* oF, __bf16* oH, int relu) {
        urd_wprep_k<<<dim3(1, Cout * KT), CP, 0, stream>>>(w, wbf, Cin, CP, KT);
        long DHW = 1L << (3 * logS);
        dim3 grid((unsigned)(DHW >> 8), (unsigned)(Cout >> 4), 2);
#define URD_CASE(cp, kt)                                                      \
        urd_conv_wmma_k<cp, kt><<<grid, 128, 0, stream>>>(                    \
            act, wbf, g, b, m, v, resid, oF, oH, Cout, CoutPad, logS, relu)
        if (KT == 27) {
            switch (CP) {
                case 32:  URD_CASE(32, 27);  break;
                case 64:  URD_CASE(64, 27);  break;
                case 96:  URD_CASE(96, 27);  break;
                default:  URD_CASE(128, 27); break;
            }
        } else {
            switch (CP) {
                case 32:  URD_CASE(32, 1);  break;
                case 64:  URD_CASE(64, 1);  break;
                case 96:  URD_CASE(96, 1);  break;
                default:  URD_CASE(128, 1); break;
            }
        }
#undef URD_CASE
    };

    for (int idx = 0; idx < 4; ++idx) {
        int i     = 3 - idx;
        int cinD  = PL[i + 1];
        int coutD = PL[i];
        int logSi = 5 - i;                 // sin = 64 >> (i+1) = 2^(5-i)
        int logSo = logSi + 1;
        long DHW  = 1L << (3 * logSo);
        long DHWi = 1L << (3 * logSi);
        int Ctot  = 2 * coutD;
        int Cpad2 = (coutD + 31) & ~31;

        const float* x = (idx == 0) ? xfinal : (outf + out_off[idx - 1]);

        // 1) x = relu(bn(x))  (f32)
        urd_bn_relu_k<<<dim3(urd_cdiv(DHWi, 256), 2 * cinD), 256, 0, stream>>>(
            x, dec[idx].g, dec[idx].b, dec[idx].m, dec[idx].v, bnx, cinD, (int)DHWi);

        // 2) encoder skip -> channels [0, coutD) of channels-last concat
        urd_copy_enc_k<<<dim3(urd_cdiv(DHW, 256), 2 * coutD), 256, 0, stream>>>(
            enc[i], xcatH, coutD, Ctot, (int)DHW);

        // 3) transposed conv 2x2x2/2 -> channels [coutD, Ctot)
        urd_convt2_k<<<dim3(urd_cdiv(DHW, 256), 2 * coutD), 256, 0, stream>>>(
            bnx, dec[idx].wt, xcatH, cinD, coutD, Ctot, coutD, logSi);

        // zero the bf16 padding channels of t1h/xbH for this stage's layout
        if (Cpad2 > coutD) {
            long ptot = 2 * DHW * (Cpad2 - coutD);
            urd_padzero_k<<<urd_cdiv(ptot, 256), 256, 0, stream>>>(t1h, coutD, Cpad2, ptot);
            urd_padzero_k<<<urd_cdiv(ptot, 256), 256, 0, stream>>>(xbH, coutD, Cpad2, ptot);
        }

        // 4) resblock 0 (fin = Ctot -> coutD, 1x1 residual via same WMMA kernel)
        const UrdBlk* b0 = &blk[idx][0];
        launch_conv(Ctot, 27, xcatH, b0->w1, Ctot, coutD, Cpad2, logSo,
                    b0->g1, b0->b1, b0->m1, b0->v1, nullptr, nullptr, t1h, 1);
        launch_conv(Ctot, 1, xcatH, b0->res_w, Ctot, coutD, Cpad2, logSo,
                    nullptr, nullptr, nullptr, nullptr, nullptr, resb, nullptr, 0);
        launch_conv(Cpad2, 27, t1h, b0->w2, coutD, coutD, Cpad2, logSo,
                    b0->g2, b0->b2, b0->m2, b0->v2, resb, xbF, xbH, 1);

        // 5) resblock 1 (identity residual), final conv writes the d_out slice
        const UrdBlk* b1 = &blk[idx][1];
        launch_conv(Cpad2, 27, xbH, b1->w1, coutD, coutD, Cpad2, logSo,
                    b1->g1, b1->b1, b1->m1, b1->v1, nullptr, nullptr, t1h, 1);
        launch_conv(Cpad2, 27, t1h, b1->w2, coutD, coutD, Cpad2, logSo,
                    b1->g2, b1->b2, b1->m2, b1->v2, xbF,
                    outf + out_off[idx], nullptr, 1);
    }
}
